// GraphAT_cellChat_24910810316939
// MI455X (gfx1250) — matrix operand order repaired and checked by reading, not compiled, not run on hardware
//
#include <hip/hip_runtime.h>
#include <hip/hip_bf16.h>

typedef __attribute__((ext_vector_type(16))) _Float16 v16h;
typedef __attribute__((ext_vector_type(8)))  _Float16 v8h;
typedef __attribute__((ext_vector_type(8)))  float    v8f;

// ---------------------------------------------------------------------------
// Weight pre-pack: f32 row-major W[K][COLS]  ->  f16 WMMA-B fragment layout.
// B 32x16 f16 fragment: lane l holds column (l&15); lanes 0-15 carry K 0..7,16..23,
// lanes 16-31 carry K 8..15,24..31 of the 32-wide chunk; 16 halves per lane.
// pos = ((chunk*CT + colTile)*32 + lane)*16 + h
// ---------------------------------------------------------------------------
template<int K, int COLS>
__global__ void pack_b(const float* __restrict__ W, _Float16* __restrict__ out) {
    constexpr int CT  = COLS / 16;
    constexpr int CH  = K / 32;
    constexpr int TOT = CH * CT * 512;
    for (int pos = threadIdx.x; pos < TOT; pos += blockDim.x) {
        int h  = pos & 15;
        int l  = (pos >> 4) & 31;
        int tt = pos >> 9;            // chunk*CT + colTile
        int ct = tt % CT;
        int ch = tt / CT;
        int k  = ch * 32 + (h >> 3) * 16 + (l >> 4) * 8 + (h & 7);
        int n  = ct * 16 + (l & 15);
        out[pos] = (_Float16)W[k * COLS + n];
    }
}

// ---------------------------------------------------------------------------
// WMMA GEMM: Out[N x COLS] = A[N x K] * Bpacked, f32 accumulate.
// One wave per 16-row tile, 8 waves per block.
// ---------------------------------------------------------------------------
template<int K, int COLS, bool AHALF>
__global__ void gemm_wmma(const void* __restrict__ Aptr,
                          const _Float16* __restrict__ Bp,
                          float* __restrict__ Out, int Nn) {
    constexpr int CT = COLS / 16;
    constexpr int CH = K / 32;
    const int lane    = threadIdx.x & 31;
    const int wave    = threadIdx.x >> 5;
    const int tileRow = blockIdx.x * 8 + wave;
    const int row0    = tileRow * 16;
    if (row0 >= Nn) return;                 // wave-uniform exit (EXEC stays full)
    const int g = lane >> 4;                // half-wave id
    const int m = lane & 15;
    int rowA = row0 + m;
    if (rowA >= Nn) rowA = Nn - 1;          // clamp loads for tail tile

    v8f acc[CT];
#pragma unroll
    for (int t = 0; t < CT; ++t)
#pragma unroll
        for (int j = 0; j < 8; ++j) acc[t][j] = 0.0f;

    const float*    Af = (const float*)Aptr;
    const _Float16* Ah = (const _Float16*)Aptr;

#pragma unroll
    for (int ch = 0; ch < CH; ++ch) {
        const int k0 = ch * 32;
        v16h a;
        if constexpr (AHALF) {
            const v8h* p0 = (const v8h*)(Ah + (size_t)rowA * K + k0 + g * 8);
            const v8h* p1 = (const v8h*)(Ah + (size_t)rowA * K + k0 + 16 + g * 8);
            v8h lo = *p0, hi = *p1;
            a = __builtin_shufflevector(lo, hi, 0,1,2,3,4,5,6,7,8,9,10,11,12,13,14,15);
        } else {
            const float4* p0 = (const float4*)(Af + (size_t)rowA * K + k0 + g * 8);
            const float4* p1 = (const float4*)(Af + (size_t)rowA * K + k0 + 16 + g * 8);
            float4 f0 = p0[0], f1 = p0[1], f2 = p1[0], f3 = p1[1];
            a[0]=(_Float16)f0.x;  a[1]=(_Float16)f0.y;  a[2]=(_Float16)f0.z;  a[3]=(_Float16)f0.w;
            a[4]=(_Float16)f1.x;  a[5]=(_Float16)f1.y;  a[6]=(_Float16)f1.z;  a[7]=(_Float16)f1.w;
            a[8]=(_Float16)f2.x;  a[9]=(_Float16)f2.y;  a[10]=(_Float16)f2.z; a[11]=(_Float16)f2.w;
            a[12]=(_Float16)f3.x; a[13]=(_Float16)f3.y; a[14]=(_Float16)f3.z; a[15]=(_Float16)f3.w;
        }
#pragma unroll
        for (int t = 0; t < CT; ++t) {
            const v16h* bp = (const v16h*)(Bp + ((size_t)(ch * CT + t) * 32 + lane) * 16);
            v16h b = *bp;
            acc[t] = __builtin_amdgcn_wmma_f32_16x16x32_f16(
                false, a, false, b, (short)0, acc[t], false, false);
        }
    }
    // C/D layout: lane holds column (lane&15) of col-tile; VGPR r -> row r + 8*g
#pragma unroll
    for (int t = 0; t < CT; ++t) {
        const int col = t * 16 + m;
#pragma unroll
        for (int r = 0; r < 8; ++r) {
            int rr = row0 + r + g * 8;
            if (rr < Nn) Out[(size_t)rr * COLS + col] = acc[t][r];
        }
    }
}

// ---------------------------------------------------------------------------
// Per-node attention coefficients: as[i]=h[i].att_src, ad[i]=h[i].att_dst
// ---------------------------------------------------------------------------
template<int C>
__global__ void alphas_k(const float* __restrict__ h, const float* __restrict__ avs,
                         const float* __restrict__ avd, float* __restrict__ as_o,
                         float* __restrict__ ad_o, int Nn) {
    int lane = threadIdx.x & 31;
    int node = blockIdx.x * 8 + (threadIdx.x >> 5);
    if (node >= Nn) return;
    float s1 = 0.f, s2 = 0.f;
    for (int c = lane; c < C; c += 32) {
        float v = h[(size_t)node * C + c];
        s1 += v * avs[c];
        s2 += v * avd[c];
    }
    for (int off = 16; off; off >>= 1) {
        s1 += __shfl_xor(s1, off, 32);
        s2 += __shfl_xor(s2, off, 32);
    }
    if (lane == 0) { as_o[node] = s1; ad_o[node] = s2; }
}

// ---------------------------------------------------------------------------
// Ordered-uint encoding so f32 max can use native u32 atomicMax.
// ---------------------------------------------------------------------------
__device__ __forceinline__ unsigned f2ord(float f) {
    unsigned u = __float_as_uint(f);
    return (u & 0x80000000u) ? ~u : (u | 0x80000000u);
}
__device__ __forceinline__ float ord2f(unsigned k) {
    return (k & 0x80000000u) ? __uint_as_float(k & 0x7fffffffu) : __uint_as_float(~k);
}

__device__ __forceinline__ void edge_sd(const int* ei, int E, int i, int& s, int& d) {
    if (i < E) { s = ei[i]; d = ei[E + i]; }
    else       { s = i - E; d = i - E; }     // appended self-loops
}

__global__ void edge_max(const int* __restrict__ ei, int E, int EN,
                         const float* __restrict__ as, const float* __restrict__ ad,
                         unsigned* __restrict__ mord) {
    int i = blockIdx.x * blockDim.x + threadIdx.x;
    if (i >= EN) return;
    int s, d; edge_sd(ei, E, i, s, d);
    float v = as[s] + ad[d];
    v = v > 0.f ? v : 0.2f * v;              // GAT leaky_relu
    atomicMax(&mord[d], f2ord(v));
}

__global__ void edge_exp(const int* __restrict__ ei, int E, int EN,
                         const float* __restrict__ as, const float* __restrict__ ad,
                         const unsigned* __restrict__ mord,
                         float* __restrict__ ex, float* __restrict__ ssum) {
    int i = blockIdx.x * blockDim.x + threadIdx.x;
    if (i >= EN) return;
    int s, d; edge_sd(ei, E, i, s, d);
    float v = as[s] + ad[d];
    v = v > 0.f ? v : 0.2f * v;
    float e = __expf(v - ord2f(mord[d]));
    ex[i] = e;
    atomicAdd(&ssum[d], e);
}

// One wave per edge: coalesced row gather of h[src], f32 atomic scatter to agg[dst]
template<int C>
__global__ void edge_agg(const int* __restrict__ ei, int E, int EN,
                         const float* __restrict__ ex, const float* __restrict__ ssum,
                         const float* __restrict__ h, float* __restrict__ agg) {
    int w    = blockIdx.x * 8 + (threadIdx.x >> 5);
    int lane = threadIdx.x & 31;
    if (w >= EN) return;
    int s, d; edge_sd(ei, E, w, s, d);
    float alpha = ex[w] / (ssum[d] + 1e-16f);
    for (int c = lane; c < C; c += 32)
        atomicAdd(&agg[(size_t)d * C + c], h[(size_t)s * C + c] * alpha);
}

// ---------------------------------------------------------------------------
// BatchNorm over nodes (64 channels). b1 cancels exactly through mean-subtract.
// ---------------------------------------------------------------------------
__global__ void bn_stats(const float* __restrict__ agg, int Nn,
                         float* __restrict__ gsum, float* __restrict__ gsq) {
    __shared__ float ls[256], lq[256];
    int c  = threadIdx.x & 63;
    int rg = threadIdx.x >> 6;               // 0..3
    float s = 0.f, q = 0.f;
    for (int row = blockIdx.x * 4 + rg; row < Nn; row += gridDim.x * 4) {
        float v = agg[(size_t)row * 64 + c];
        s += v; q += v * v;
    }
    ls[threadIdx.x] = s; lq[threadIdx.x] = q;
    __syncthreads();
    if (rg == 0) {
        s = ls[c] + ls[64 + c] + ls[128 + c] + ls[192 + c];
        q = lq[c] + lq[64 + c] + lq[128 + c] + lq[192 + c];
        atomicAdd(&gsum[c], s);
        atomicAdd(&gsq[c], q);
    }
}

__global__ void bn_final(const float* gsum, const float* gsq,
                         const float* gamma, const float* beta, int Nn,
                         float* scale, float* shift) {
    int c = threadIdx.x;
    if (c >= 64) return;
    float inv = 1.0f / (float)Nn;
    float mu  = gsum[c] * inv;
    float var = gsq[c] * inv - mu * mu;
    float sc  = gamma[c] * rsqrtf(var + 1e-5f);
    scale[c] = sc;
    shift[c] = beta[c] - mu * sc;
}

// BN + leaky_relu(0.01), store f16 for layer-2 WMMA A operand
__global__ void bn_apply(const float* __restrict__ agg, const float* __restrict__ scale,
                         const float* __restrict__ shift, _Float16* __restrict__ hid,
                         int total) {
    int i = blockIdx.x * blockDim.x + threadIdx.x;
    if (i >= total) return;
    int c = i & 63;
    float t = agg[i] * scale[c] + shift[c];
    t = t > 0.f ? t : 0.01f * t;
    hid[i] = (_Float16)t;
}

// Final row softmax over 32 channels (+ b2), one wave per node
__global__ void softmax_out(const float* __restrict__ agg, const float* __restrict__ b2,
                            float* __restrict__ out, int Nn) {
    int lane = threadIdx.x & 31;
    int node = blockIdx.x * 8 + (threadIdx.x >> 5);
    if (node >= Nn) return;
    float v = agg[(size_t)node * 32 + lane] + b2[lane];
    float m = v;
    for (int off = 16; off; off >>= 1) m = fmaxf(m, __shfl_xor(m, off, 32));
    float e = __expf(v - m);
    float s = e;
    for (int off = 16; off; off >>= 1) s += __shfl_xor(s, off, 32);
    out[(size_t)node * 32 + lane] = e / s;
}

// ---------------------------------------------------------------------------
extern "C" void kernel_launch(void* const* d_in, const int* in_sizes, int n_in,
                              void* d_out, int out_size, void* d_ws, size_t ws_size,
                              hipStream_t stream) {
    const float* x      = (const float*)d_in[0];
    const int*   ei     = (const int*)d_in[1];
    const float* W1     = (const float*)d_in[2];
    const float* att_s1 = (const float*)d_in[3];
    const float* att_d1 = (const float*)d_in[4];
    // d_in[5] = b1: provably cancels through BatchNorm mean-subtraction
    const float* gamma  = (const float*)d_in[6];
    const float* beta   = (const float*)d_in[7];
    const float* W2     = (const float*)d_in[8];
    const float* att_s2 = (const float*)d_in[9];
    const float* att_d2 = (const float*)d_in[10];
    const float* b2     = (const float*)d_in[11];

    const int Nn = in_sizes[0] / 256;
    const int E  = in_sizes[1] / 2;
    const int EN = E + Nn;

    char* p = (char*)d_ws;
    auto alloc = [&](size_t bytes) -> void* {
        void* r = (void*)p;
        p += (bytes + 255) & ~(size_t)255;
        return r;
    };
    _Float16* W1h  = (_Float16*)alloc(256 * 64 * sizeof(_Float16));
    _Float16* W2h  = (_Float16*)alloc(64 * 32 * sizeof(_Float16));
    float*    h1   = (float*)alloc((size_t)Nn * 64 * 4);
    float*    agg1 = (float*)alloc((size_t)Nn * 64 * 4);
    _Float16* hid  = (_Float16*)alloc((size_t)Nn * 64 * 2);
    float*    h2   = (float*)alloc((size_t)Nn * 32 * 4);
    float*    agg2 = (float*)alloc((size_t)Nn * 32 * 4);
    float*    as1  = (float*)alloc((size_t)Nn * 4);
    float*    ad1  = (float*)alloc((size_t)Nn * 4);
    float*    as2  = (float*)alloc((size_t)Nn * 4);
    float*    ad2  = (float*)alloc((size_t)Nn * 4);
    unsigned* mo1  = (unsigned*)alloc((size_t)Nn * 4);
    float*    ss1  = (float*)alloc((size_t)Nn * 4);
    unsigned* mo2  = (unsigned*)alloc((size_t)Nn * 4);
    float*    ss2  = (float*)alloc((size_t)Nn * 4);
    float*    ex   = (float*)alloc((size_t)EN * 4);
    float*    gsum = (float*)alloc(64 * 4);
    float*    gsq  = (float*)alloc(64 * 4);
    float*    bsc  = (float*)alloc(64 * 4);
    float*    bsh  = (float*)alloc(64 * 4);

    hipMemsetAsync(agg1, 0, (size_t)Nn * 64 * 4, stream);
    hipMemsetAsync(agg2, 0, (size_t)Nn * 32 * 4, stream);
    hipMemsetAsync(mo1,  0, (size_t)Nn * 4, stream);   // ord(-inf) sentinel
    hipMemsetAsync(ss1,  0, (size_t)Nn * 4, stream);
    hipMemsetAsync(mo2,  0, (size_t)Nn * 4, stream);
    hipMemsetAsync(ss2,  0, (size_t)Nn * 4, stream);
    hipMemsetAsync(gsum, 0, 64 * 4, stream);
    hipMemsetAsync(gsq,  0, 64 * 4, stream);

    pack_b<256, 64><<<1, 256, 0, stream>>>(W1, W1h);
    pack_b<64, 32><<<1, 256, 0, stream>>>(W2, W2h);

    const int gemmBlocks = (Nn + 127) / 128;
    const int edgeBlocks = (EN + 255) / 256;
    const int waveBlocks = (EN + 7) / 8;
    const int nodeBlocks = (Nn + 7) / 8;

    // ---- layer 1 ----
    gemm_wmma<256, 64, false><<<gemmBlocks, 256, 0, stream>>>(x, W1h, h1, Nn);
    alphas_k<64><<<nodeBlocks, 256, 0, stream>>>(h1, att_s1, att_d1, as1, ad1, Nn);
    edge_max<<<edgeBlocks, 256, 0, stream>>>(ei, E, EN, as1, ad1, mo1);
    edge_exp<<<edgeBlocks, 256, 0, stream>>>(ei, E, EN, as1, ad1, mo1, ex, ss1);
    edge_agg<64><<<waveBlocks, 256, 0, stream>>>(ei, E, EN, ex, ss1, h1, agg1);

    // ---- BN + act ----
    bn_stats<<<256, 256, 0, stream>>>(agg1, Nn, gsum, gsq);
    bn_final<<<1, 64, 0, stream>>>(gsum, gsq, gamma, beta, Nn, bsc, bsh);
    bn_apply<<<((Nn * 64) + 255) / 256, 256, 0, stream>>>(agg1, bsc, bsh, hid, Nn * 64);

    // ---- layer 2 ----
    gemm_wmma<64, 32, true><<<gemmBlocks, 256, 0, stream>>>(hid, W2h, h2, Nn);
    alphas_k<32><<<nodeBlocks, 256, 0, stream>>>(h2, att_s2, att_d2, as2, ad2, Nn);
    edge_max<<<edgeBlocks, 256, 0, stream>>>(ei, E, EN, as2, ad2, mo2);
    edge_exp<<<edgeBlocks, 256, 0, stream>>>(ei, E, EN, as2, ad2, mo2, ex, ss2);
    edge_agg<32><<<waveBlocks, 256, 0, stream>>>(ei, E, EN, ex, ss2, h2, agg2);

    softmax_out<<<nodeBlocks, 256, 0, stream>>>(agg2, b2, (float*)d_out, Nn);
}
